// BiLSTM_45612552684166
// MI455X (gfx1250) — matrix operand order repaired and checked by reading, not compiled, not run on hardware
//
#include <hip/hip_runtime.h>
#include <hip/hip_bf16.h>

typedef __attribute__((ext_vector_type(16))) __bf16 v16bf;
typedef __attribute__((ext_vector_type(8)))  __bf16 v8bf;
typedef __attribute__((ext_vector_type(8)))  float  v8f;

#define TSEQ 2048
#define HD   256

__device__ __forceinline__ unsigned short f2bf(float f) {
  unsigned int u = __float_as_uint(f);
  unsigned int r = (u + 0x7FFFu + ((u >> 16) & 1u)) >> 16;   // round-to-nearest-even
  return (unsigned short)r;
}

// A fragment: per-lane 16 contiguous K values (matches B fragment order)
__device__ __forceinline__ v16bf load_afrag(const __bf16* p) {
  v8bf x0 = *(const v8bf*)(p);
  v8bf x1 = *(const v8bf*)(p + 8);
  return __builtin_shufflevector(x0, x1, 0,1,2,3,4,5,6,7,8,9,10,11,12,13,14,15);
}
// B fragment: per-lane 16 contiguous K values (matches A fragment order)
__device__ __forceinline__ v16bf load_bfrag(const __bf16* p) {
  v8bf x0 = *(const v8bf*)(p);
  v8bf x1 = *(const v8bf*)(p + 8);
  return __builtin_shufflevector(x0, x1, 0,1,2,3,4,5,6,7,8,9,10,11,12,13,14,15);
}

// ---------------- prep kernels ----------------
__global__ void k_f32_to_bf16(const float* __restrict__ in,
                              unsigned short* __restrict__ out, int n) {
  int i = blockIdx.x * blockDim.x + threadIdx.x;
  if (i < n) out[i] = f2bf(in[i]);
}

__global__ void k_bias_sum(const float* __restrict__ a, const float* __restrict__ b,
                           float* __restrict__ o, int n) {
  int i = blockIdx.x * blockDim.x + threadIdx.x;
  if (i < n) o[i] = a[i] + b[i];
}

// Whh [2][1024][256] (fp32) -> packed bf16 pairs WhhP [2][128][1024] (u32):
//   WhhP[d][kk][j] = { bf16(Whh[d][j][2kk+1]) , bf16(Whh[d][j][2kk]) }
__global__ void k_pack_whh(const float* __restrict__ Whh, unsigned int* __restrict__ WhhP) {
  int idx = blockIdx.x * blockDim.x + threadIdx.x;           // 2*128*1024
  if (idx >= 2 * 128 * 1024) return;
  int d  = idx >> 17;
  int rest = idx & 131071;
  int kk = rest >> 10;
  int j  = rest & 1023;
  const float* src = Whh + (size_t)d * 262144 + (size_t)j * 256 + 2 * kk;
  unsigned int lo = f2bf(src[0]);
  unsigned int hi = f2bf(src[1]);
  WhhP[idx] = lo | (hi << 16);
}

// Wsum[n][k] = W3[n][k] + W3[n][512+k]   (symmetrized pair weight), bf16
__global__ void k_make_wsum(const float* __restrict__ W3, unsigned short* __restrict__ Wsum) {
  int idx = blockIdx.x * blockDim.x + threadIdx.x;           // 1024*512
  if (idx >= 1024 * 512) return;
  int n = idx >> 9, k = idx & 511;
  Wsum[idx] = f2bf(W3[n * 1024 + k] + W3[n * 1024 + 512 + k]);
}

// pre0[chain][dir][t][j] = x[t] . Wih0[dir][j] + bih0+bhh0   (K=22, VALU)
__global__ void k_pre0(const float* __restrict__ vr, const float* __restrict__ vl,
                       const float* __restrict__ Wih0,
                       const float* __restrict__ bih0, const float* __restrict__ bhh0,
                       float* __restrict__ pre_r, float* __restrict__ pre_l) {
  int idx = blockIdx.x * blockDim.x + threadIdx.x;           // 2*2*2048*1024
  int j = idx & 1023;
  int t = (idx >> 10) & 2047;
  int d = (idx >> 21) & 1;
  int chain = idx >> 22;
  const float* x = (chain ? vl : vr) + t * 22;
  const float* w = Wih0 + (size_t)(d * 1024 + j) * 22;
  float acc = bih0[d * 1024 + j] + bhh0[d * 1024 + j];
#pragma unroll
  for (int k = 0; k < 22; ++k) acc += x[k] * w[k];
  float* pre = chain ? pre_l : pre_r;
  pre[((size_t)d * TSEQ + t) * 1024 + j] = acc;
}

// ---------------- persistent bidirectional LSTM scan ----------------
// 4 blocks = {chain r/l} x {fwd/bwd}; 1024 threads; h kept in LDS; packed-bf16
// Whh streamed from L2 (coalesced u32 per thread per packed k-pair).
__global__ __launch_bounds__(1024)
void k_lstm_scan(const float* __restrict__ pre_r, const float* __restrict__ pre_l,
                 const unsigned int* __restrict__ WhhP,   // [2][128][1024] packed bf16x2
                 float* __restrict__ out_r, float* __restrict__ out_l) {
  int chain = blockIdx.x >> 1;
  int dir   = blockIdx.x & 1;
  const float* pre = (chain ? pre_l : pre_r) + (size_t)dir * TSEQ * 1024;
  float* out = chain ? out_l : out_r;
  const unsigned int* W = WhhP + (size_t)dir * 128 * 1024;
  int j = threadIdx.x;

  __shared__ float h_lds[HD];
  __shared__ float g_lds[1024];

  float c = 0.f;
  if (j < HD) h_lds[j] = 0.f;
  __syncthreads();

  for (int s = 0; s < TSEQ; ++s) {
    int t = dir ? (TSEQ - 1 - s) : s;
    float acc = pre[(size_t)t * 1024 + j];
#pragma unroll 8
    for (int kk = 0; kk < 128; ++kk) {
      unsigned int w = W[kk * 1024 + j];
      float w0 = __uint_as_float(w << 16);
      float w1 = __uint_as_float(w & 0xffff0000u);
      acc += h_lds[2 * kk]     * w0;
      acc += h_lds[2 * kk + 1] * w1;
    }
    g_lds[j] = acc;
    __syncthreads();
    if (j < HD) {
      float gi = g_lds[j];
      float gf = g_lds[j + 256];
      float gg = g_lds[j + 512];
      float go = g_lds[j + 768];
      float i_ = 1.f / (1.f + __expf(-gi));
      float f_ = 1.f / (1.f + __expf(-gf));
      float g_ = tanhf(gg);
      float o_ = 1.f / (1.f + __expf(-go));
      c = f_ * c + i_ * g_;
      float h = o_ * tanhf(c);
      h_lds[j] = h;
      out[(size_t)t * 512 + dir * 256 + j] = h;      // concat(fwd, bwd)
    }
    __syncthreads();
  }
}

// ---------------- bf16 WMMA GEMM:  C[M,N] = act(A[M,K] @ B[N,K]^T + bias) ----------------
// 16x64 tile per wave (one A fragment amortized over 4 WMMAs), 4 waves/block.
// Manually software-pipelined: next K-step fragments are prefetched into a
// second live register set before the WMMA group consumes the current set, so
// the allocator cannot coalesce destinations and the WMMAs wait only on loads
// issued a full loop body earlier.
__global__ __launch_bounds__(128)
void k_gemm_bf16(const unsigned short* __restrict__ Aus, const unsigned short* __restrict__ Bus,
                 const float* __restrict__ bias, float* __restrict__ C,
                 int M, int N, int K, int relu) {
  const __bf16* A = (const __bf16*)Aus;
  const __bf16* B = (const __bf16*)Bus;
  int wave = threadIdx.x >> 5;
  int lane = threadIdx.x & 31;
  int hi   = lane >> 4;
  int r    = lane & 15;
  int m0 = blockIdx.x * 16;
  int n0 = blockIdx.y * 256 + wave * 64;
  if (m0 >= M || n0 >= N) return;                   // wave-uniform; EXEC stays all-ones

  const __bf16* arow  = A + (size_t)(m0 + r) * K + (hi ? 16 : 0);
  const __bf16* brow0 = B + (size_t)(n0 +  0 + r) * K + (hi ? 16 : 0);
  const __bf16* brow1 = B + (size_t)(n0 + 16 + r) * K + (hi ? 16 : 0);
  const __bf16* brow2 = B + (size_t)(n0 + 32 + r) * K + (hi ? 16 : 0);
  const __bf16* brow3 = B + (size_t)(n0 + 48 + r) * K + (hi ? 16 : 0);

  v8f acc0 = {}, acc1 = {}, acc2 = {}, acc3 = {};

  // prologue: fragments for k = 0
  v16bf a  = load_afrag(arow);
  v16bf b0 = load_bfrag(brow0);
  v16bf b1 = load_bfrag(brow1);
  v16bf b2 = load_bfrag(brow2);
  v16bf b3 = load_bfrag(brow3);

  for (int k = 32; k < K; k += 32) {
    // prefetch next K-step into a distinct live set
    v16bf an  = load_afrag(arow + k);
    v16bf bn0 = load_bfrag(brow0 + k);
    v16bf bn1 = load_bfrag(brow1 + k);
    v16bf bn2 = load_bfrag(brow2 + k);
    v16bf bn3 = load_bfrag(brow3 + k);
    // consume current set while next-set loads are in flight
    acc0 = __builtin_amdgcn_wmma_f32_16x16x32_bf16(false, a, false, b0, (short)0, acc0, false, false);
    acc1 = __builtin_amdgcn_wmma_f32_16x16x32_bf16(false, a, false, b1, (short)0, acc1, false, false);
    acc2 = __builtin_amdgcn_wmma_f32_16x16x32_bf16(false, a, false, b2, (short)0, acc2, false, false);
    acc3 = __builtin_amdgcn_wmma_f32_16x16x32_bf16(false, a, false, b3, (short)0, acc3, false, false);
    a = an; b0 = bn0; b1 = bn1; b2 = bn2; b3 = bn3;
  }
  // epilogue: last K-step
  acc0 = __builtin_amdgcn_wmma_f32_16x16x32_bf16(false, a, false, b0, (short)0, acc0, false, false);
  acc1 = __builtin_amdgcn_wmma_f32_16x16x32_bf16(false, a, false, b1, (short)0, acc1, false, false);
  acc2 = __builtin_amdgcn_wmma_f32_16x16x32_bf16(false, a, false, b2, (short)0, acc2, false, false);
  acc3 = __builtin_amdgcn_wmma_f32_16x16x32_bf16(false, a, false, b3, (short)0, acc3, false, false);

  // C/D layout: row = vgpr + 8*hi, col = n-subtile base + r
#pragma unroll
  for (int tsub = 0; tsub < 4; ++tsub) {
    const v8f& acc = tsub == 0 ? acc0 : tsub == 1 ? acc1 : tsub == 2 ? acc2 : acc3;
    int n = n0 + 16 * tsub + r;
    float bn = bias ? bias[n] : 0.f;
#pragma unroll
    for (int v = 0; v < 8; ++v) {
      float x = acc[v] + bn;
      if (relu) x = fmaxf(x, 0.f);
      C[(size_t)(m0 + v + hi * 8) * N + n] = x;
    }
  }
}

// ---------------- pair scorer: log_softmax(Wout @ relu(0.5*(S_r[r]+S_l[l]) + b3) + bout) ----
// Each lane owns a contiguous 32-element (128B) segment -> float4 loads.
__global__ __launch_bounds__(256)
void k_pair(const float* __restrict__ Sr, const float* __restrict__ Sl,
            const int* __restrict__ pr, const int* __restrict__ pl,
            const float* __restrict__ b3, const float* __restrict__ Wout,
            const float* __restrict__ bout, float* __restrict__ out, int npairs) {
  __shared__ float sb3[1024], sw0[1024], sw1[1024];
  for (int i = threadIdx.x; i < 1024; i += 256) {
    sb3[i] = b3[i];
    sw0[i] = Wout[i];
    sw1[i] = Wout[1024 + i];
  }
  __syncthreads();

  int wave = threadIdx.x >> 5, lane = threadIdx.x & 31;
  int p = blockIdx.x * 8 + wave;
  if (p >= npairs) return;
  int base = lane * 32;
  const float4* ar = (const float4*)(Sr + (size_t)pr[p] * 1024 + base);
  const float4* al = (const float4*)(Sl + (size_t)pl[p] * 1024 + base);
  const float4* vb = (const float4*)(sb3 + base);
  const float4* v0 = (const float4*)(sw0 + base);
  const float4* v1 = (const float4*)(sw1 + base);
  __builtin_prefetch(ar, 0, 1);                     // global_prefetch_b8
  __builtin_prefetch(al, 0, 1);

  float a0 = 0.f, a1 = 0.f;
#pragma unroll
  for (int i = 0; i < 8; ++i) {
    float4 xr = ar[i], xl = al[i], xb = vb[i], w0 = v0[i], w1 = v1[i];
    float x;
    x = fmaxf(0.5f * (xr.x + xl.x) + xb.x, 0.f); a0 += x * w0.x; a1 += x * w1.x;
    x = fmaxf(0.5f * (xr.y + xl.y) + xb.y, 0.f); a0 += x * w0.y; a1 += x * w1.y;
    x = fmaxf(0.5f * (xr.z + xl.z) + xb.z, 0.f); a0 += x * w0.z; a1 += x * w1.z;
    x = fmaxf(0.5f * (xr.w + xl.w) + xb.w, 0.f); a0 += x * w0.w; a1 += x * w1.w;
  }
#pragma unroll
  for (int off = 16; off; off >>= 1) {
    a0 += __shfl_xor(a0, off, 32);
    a1 += __shfl_xor(a1, off, 32);
  }
  if (lane == 0) {
    float l0 = a0 + bout[0], l1 = a1 + bout[1];
    float m = fmaxf(l0, l1);
    float lse = m + logf(__expf(l0 - m) + __expf(l1 - m));
    out[(size_t)p * 2 + 0] = l0 - lse;
    out[(size_t)p * 2 + 1] = l1 - lse;
  }
}

// ---------------- orchestration ----------------
extern "C" void kernel_launch(void* const* d_in, const int* in_sizes, int n_in,
                              void* d_out, int out_size, void* d_ws, size_t ws_size,
                              hipStream_t stream) {
  const float* v_r   = (const float*)d_in[0];
  const float* v_l   = (const float*)d_in[1];
  const int*   pr    = (const int*)  d_in[2];
  const int*   pl    = (const int*)  d_in[3];
  const float* Wih0  = (const float*)d_in[4];
  const float* Whh0  = (const float*)d_in[5];
  const float* bih0  = (const float*)d_in[6];
  const float* bhh0  = (const float*)d_in[7];
  const float* Wih1  = (const float*)d_in[8];
  const float* Whh1  = (const float*)d_in[9];
  const float* bih1  = (const float*)d_in[10];
  const float* bhh1  = (const float*)d_in[11];
  const float* W1    = (const float*)d_in[12];
  const float* b1    = (const float*)d_in[13];
  const float* W2    = (const float*)d_in[14];
  const float* b2    = (const float*)d_in[15];
  const float* W3    = (const float*)d_in[16];
  const float* b3    = (const float*)d_in[17];
  const float* Wout  = (const float*)d_in[18];
  const float* bout  = (const float*)d_in[19];
  float* out = (float*)d_out;
  int npairs = in_sizes[2];

  char* ws = (char*)d_ws;
  size_t off = 0;
  auto alloc = [&](size_t bytes) -> void* {
    void* p = ws + off;
    off += (bytes + 255) & ~(size_t)255;
    return p;
  };

  unsigned int* WhhP0 = (unsigned int*)alloc((size_t)2 * 128 * 1024 * 4);
  unsigned int* WhhP1 = (unsigned int*)alloc((size_t)2 * 128 * 1024 * 4);
  float* bsum1 = (float*)alloc((size_t)2 * 1024 * 4);
  unsigned short* Wih1b = (unsigned short*)alloc((size_t)2 * 1024 * 512 * 2);
  unsigned short* W1b   = (unsigned short*)alloc((size_t)1024 * 512 * 2);
  unsigned short* W2b   = (unsigned short*)alloc((size_t)512 * 1024 * 2);
  unsigned short* Wsumb = (unsigned short*)alloc((size_t)1024 * 512 * 2);
  float* pre[2];   for (int c = 0; c < 2; ++c) pre[c]   = (float*)alloc((size_t)2 * TSEQ * 1024 * 4);
  float* h0[2];    for (int c = 0; c < 2; ++c) h0[c]    = (float*)alloc((size_t)TSEQ * 512 * 4);
  float* h1[2];    for (int c = 0; c < 2; ++c) h1[c]    = (float*)alloc((size_t)TSEQ * 512 * 4);
  unsigned short* hbf[2];   for (int c = 0; c < 2; ++c) hbf[c]   = (unsigned short*)alloc((size_t)TSEQ * 512 * 2);
  float* mlp1[2];  for (int c = 0; c < 2; ++c) mlp1[c]  = (float*)alloc((size_t)TSEQ * 1024 * 4);
  unsigned short* mlp1b[2]; for (int c = 0; c < 2; ++c) mlp1b[c] = (unsigned short*)alloc((size_t)TSEQ * 1024 * 2);
  float* enc[2];   for (int c = 0; c < 2; ++c) enc[c]   = (float*)alloc((size_t)TSEQ * 512 * 4);
  float* S[2];     for (int c = 0; c < 2; ++c) S[c]     = (float*)alloc((size_t)TSEQ * 1024 * 4);
  (void)ws_size;

  // weight prep
  k_pack_whh<<<1024, 256, 0, stream>>>(Whh0, WhhP0);
  k_pack_whh<<<1024, 256, 0, stream>>>(Whh1, WhhP1);
  k_bias_sum<<<8, 256, 0, stream>>>(bih1, bhh1, bsum1, 2048);
  k_f32_to_bf16<<<4096, 256, 0, stream>>>(Wih1, Wih1b, 2 * 1024 * 512);
  k_f32_to_bf16<<<2048, 256, 0, stream>>>(W1, W1b, 1024 * 512);
  k_f32_to_bf16<<<2048, 256, 0, stream>>>(W2, W2b, 512 * 1024);
  k_make_wsum<<<2048, 256, 0, stream>>>(W3, Wsumb);

  // LSTM layer 0
  k_pre0<<<32768, 256, 0, stream>>>(v_r, v_l, Wih0, bih0, bhh0, pre[0], pre[1]);
  k_lstm_scan<<<4, 1024, 0, stream>>>(pre[0], pre[1], WhhP0, h0[0], h0[1]);

  // LSTM layer 1 input projection (WMMA) then scan
  for (int c = 0; c < 2; ++c)
    k_f32_to_bf16<<<4096, 256, 0, stream>>>(h0[c], hbf[c], TSEQ * 512);
  for (int c = 0; c < 2; ++c)
    for (int d = 0; d < 2; ++d)
      k_gemm_bf16<<<dim3(128, 4), 128, 0, stream>>>(hbf[c], Wih1b + (size_t)d * 1024 * 512,
                                                    bsum1 + d * 1024,
                                                    pre[c] + (size_t)d * TSEQ * 1024,
                                                    TSEQ, 1024, 512, 0);
  k_lstm_scan<<<4, 1024, 0, stream>>>(pre[0], pre[1], WhhP1, h1[0], h1[1]);

  // MLP encoder (WMMA)
  for (int c = 0; c < 2; ++c) {
    k_f32_to_bf16<<<4096, 256, 0, stream>>>(h1[c], hbf[c], TSEQ * 512);
    k_gemm_bf16<<<dim3(128, 4), 128, 0, stream>>>(hbf[c], W1b, b1, mlp1[c], TSEQ, 1024, 512, 1);
    k_f32_to_bf16<<<8192, 256, 0, stream>>>(mlp1[c], mlp1b[c], TSEQ * 1024);
    k_gemm_bf16<<<dim3(128, 2), 128, 0, stream>>>(mlp1b[c], W2b, b2, enc[c], TSEQ, 512, 1024, 1);
    // symmetrized pair table: S = enc @ (W3a+W3b)^T  (b3 folded into the pair kernel)
    k_f32_to_bf16<<<4096, 256, 0, stream>>>(enc[c], hbf[c], TSEQ * 512);
    k_gemm_bf16<<<dim3(128, 4), 128, 0, stream>>>(hbf[c], Wsumb, nullptr, S[c], TSEQ, 1024, 512, 0);
  }

  // pairwise gather + score + log_softmax
  k_pair<<<(npairs + 7) / 8, 256, 0, stream>>>(S[0], S[1], pr, pl, b3, Wout, bout, out, npairs);
}
